// MHAttention_19310172963168
// MI455X (gfx1250) — compile-verified, matrix operand order checked by reading, not run on hardware
//
#include <hip/hip_runtime.h>
#include <hip/hip_bf16.h>
#include <stdint.h>

// ---------------------------------------------------------------------------
// Types matching CDNA5 WMMA builtin signatures
// ---------------------------------------------------------------------------
typedef __bf16        v16bf __attribute__((ext_vector_type(16)));
typedef float         v8f   __attribute__((ext_vector_type(8)));
typedef unsigned int  u32v4 __attribute__((ext_vector_type(4)));
typedef unsigned int  u32x4 __attribute__((ext_vector_type(4)));
typedef int           i32x8 __attribute__((ext_vector_type(8)));
typedef int           i32x4 __attribute__((ext_vector_type(4)));

#define B_  8
#define T_  2048
#define C_  512
#define H_  512

// Padded LDS row strides (elements). 1040B and 4112B row strides are
// == 4 dwords mod 64 banks -> ds_load_b128 A-fragment reads are conflict-free.
#define QROW 520     // 512 + 8
#define PROW 2056    // 2048 + 8

union Frag {
  u32v4 q[2];   // two 16B chunks: K = k0+8*sel .. +7  and  k0+16+8*sel .. +7
  v16bf b;
};

__device__ __forceinline__ unsigned short f2bf(float x) {
  unsigned u = __float_as_uint(x);
  return (unsigned short)((u + 0x7FFFu + ((u >> 16) & 1u)) >> 16);  // RNE
}
__device__ __forceinline__ float hmax16(float v) {
  #pragma unroll
  for (int m = 1; m < 16; m <<= 1) v = fmaxf(v, __shfl_xor(v, m, 32));
  return v;
}
__device__ __forceinline__ float hsum16(float v) {
  #pragma unroll
  for (int m = 1; m < 16; m <<= 1) v += __shfl_xor(v, m, 32);
  return v;
}
// exp-transform two packed bf16 scores -> two packed bf16 probabilities
__device__ __forceinline__ unsigned expw(unsigned w, float m, float rl) {
  float a = __uint_as_float(w << 16);
  float b = __uint_as_float(w & 0xFFFF0000u);
  float pa = __expf(a - m) * rl;
  float pb = __expf(b - m) * rl;
  return (unsigned)f2bf(pa) | ((unsigned)f2bf(pb) << 16);
}

// ---------------------------------------------------------------------------
// Kernel 1: fp32 -> bf16 conversion (x and the three weights)
// ---------------------------------------------------------------------------
__global__ __launch_bounds__(256)
void cvt_f32_bf16(const float* __restrict__ s, unsigned short* __restrict__ d, int n) {
  int i = blockIdx.x * blockDim.x + threadIdx.x;
  if (i < n) d[i] = f2bf(s[i]);
}

// ---------------------------------------------------------------------------
// Kernel 2: QKV projection.  q,k row-major bf16 [B*T, H]; v stored transposed
// vT[b][h][t] so it is a ready-made WMMA B-operand for P@V.
// grid = (256 row-tiles, 8 col-chunks, 3 weights), block = 128 (4 waves)
// ---------------------------------------------------------------------------
__global__ __launch_bounds__(128)
void qkv_wmma(const unsigned short* __restrict__ xb,
              const unsigned short* __restrict__ wqb,
              const unsigned short* __restrict__ wkb,
              const unsigned short* __restrict__ wvb,
              unsigned short* __restrict__ qb,
              unsigned short* __restrict__ kb,
              unsigned short* __restrict__ vt) {
  const int wave = threadIdx.x >> 5;
  const int lane = threadIdx.x & 31;
  const int lsel = lane >> 4;        // which K half-pattern this lane holds
  const int l15  = lane & 15;
  const int m0 = blockIdx.x * 64 + wave * 16;
  const int n0 = blockIdx.y * 64;
  const int z  = blockIdx.z;
  const unsigned short* W = (z == 0) ? wqb : (z == 1) ? wkb : wvb;

  const v8f z8 = {0.f,0.f,0.f,0.f,0.f,0.f,0.f,0.f};
  v8f acc[4];
  #pragma unroll
  for (int f = 0; f < 4; ++f) acc[f] = z8;

  const size_t arow = (size_t)(m0 + l15) * C_ + 8 * lsel;
  for (int k0 = 0; k0 < C_; k0 += 32) {
    Frag a;
    const u32v4* ap = (const u32v4*)(xb + arow + k0);
    a.q[0] = ap[0];
    a.q[1] = ap[2];
    #pragma unroll
    for (int f = 0; f < 4; ++f) {
      Frag bb;
      const u32v4* bp = (const u32v4*)(W + (size_t)(n0 + 16 * f + l15) * C_ + k0 + 8 * lsel);
      bb.q[0] = bp[0];
      bb.q[1] = bp[2];
      acc[f] = __builtin_amdgcn_wmma_f32_16x16x32_bf16(false, a.b, false, bb.b,
                                                       (short)0, acc[f], false, false);
    }
  }

  if (z < 2) {
    unsigned short* dst = (z == 0) ? qb : kb;
    const int mb = m0 + 8 * lsel;                       // D layout: M = v + 8*lsel
    #pragma unroll
    for (int f = 0; f < 4; ++f)
      #pragma unroll
      for (int v = 0; v < 8; ++v)
        dst[(size_t)(mb + v) * H_ + n0 + 16 * f + l15] = f2bf(acc[f][v]);
  } else {
    // transposed store: 8 consecutive t per lane -> one b128 store per tile
    const int bidx = m0 >> 11;
    const int t    = m0 & (T_ - 1);
    #pragma unroll
    for (int f = 0; f < 4; ++f) {
      u32v4 pk;
      #pragma unroll
      for (int p = 0; p < 4; ++p)
        pk[p] = (unsigned)f2bf(acc[f][2 * p]) | ((unsigned)f2bf(acc[f][2 * p + 1]) << 16);
      size_t off = (((size_t)(bidx * H_ + n0 + 16 * f + l15)) << 11) + t + 8 * lsel;
      *(u32v4*)(vt + off) = pk;
    }
  }
}

// ---------------------------------------------------------------------------
// Kernel 3: fused causal attention.  One 8-wave WG per (batch, 16-row tile).
// Q tile loaded to LDS by the Tensor Data Mover (with LDS padding).
// Stage A: waves split the causal s-range, WMMA scores -> LDS bf16 + online
//          per-wave softmax stats.  Stage B: each wave owns a 64-wide h-chunk,
//          exp-transforms LDS scores into WMMA A-fragments, accumulates O @ V.
// ---------------------------------------------------------------------------
__global__ __launch_bounds__(256)
void attn_wmma(const unsigned short* __restrict__ qb,
               const unsigned short* __restrict__ kb,
               const unsigned short* __restrict__ vt,
               float* __restrict__ out) {
  extern __shared__ char smem[];
  unsigned short* sh_q = (unsigned short*)smem;                       // [16][QROW]
  unsigned short* sh_p = (unsigned short*)(smem + 16 * QROW * 2);     // [16][PROW]
  float* sh_m  = (float*)(smem + 16 * QROW * 2 + 16 * PROW * 2);      // [8][16]
  float* sh_l  = sh_m + 128;                                          // [8][16]
  float* sh_mf = sh_l + 128;                                          // [16]
  float* sh_lr = sh_mf + 16;                                          // [16]

  const int tTile = blockIdx.x;
  const int b     = blockIdx.y;
  const int t0    = tTile * 16;
  const int wave  = threadIdx.x >> 5;
  const int lane  = threadIdx.x & 31;
  const int lsel  = lane >> 4;
  const int l15   = lane & 15;
  const int rowbase = b * T_ + t0;
  const float scale = 0.04419417382415922f;   // 512^-0.5
  const v8f z8 = {0.f,0.f,0.f,0.f,0.f,0.f,0.f,0.f};

  // ---- TDM: DMA the 16x512 bf16 Q tile into LDS, padded 16B per 1024B row ----
  if (wave == 0) {
    unsigned lds_off = (unsigned)(size_t)sh_q;
    unsigned long long ga = (unsigned long long)(size_t)(qb + (size_t)rowbase * C_);
    u32x4 g0;
    g0[0] = 1u;                                              // count = 1
    g0[1] = lds_off;                                         // lds_addr
    g0[2] = (unsigned)ga;                                    // global_addr lo
    g0[3] = ((unsigned)(ga >> 32) & 0x01FFFFFFu) | (2u << 30); // addr hi | type=2
    i32x8 g1;
    g1[0] = (1 << 16) | (1 << 20) | (7 << 22) | (3 << 25);   // 2B elems, pad 4dw/256dw
    g1[1] = (int)(512u << 16);                               // tensor_dim0 = 512
    g1[2] = (int)((16384u & 0xFFFFu) << 16);                 // tensor_dim1 = 16384
    g1[3] = (int)(512u << 16);                               // tile_dim0 = 512
    g1[4] = 16;                                              // tile_dim1 = 16
    g1[5] = 512;                                             // tensor_dim0_stride
    g1[6] = 0;
    g1[7] = 0;
    i32x4 z4i = {0, 0, 0, 0};
#if defined(__clang_major__) && __clang_major__ >= 23
    i32x8 zz8 = {0, 0, 0, 0, 0, 0, 0, 0};
    __builtin_amdgcn_tensor_load_to_lds(g0, g1, z4i, z4i, zz8, 0);
#else
    __builtin_amdgcn_tensor_load_to_lds(g0, g1, z4i, z4i, 0);
#endif
    __builtin_amdgcn_s_wait_tensorcnt(0);
  }
  __syncthreads();

  // ---- Stage A: scores + per-wave online softmax stats ----
  const int nblk = (tTile + 2) >> 1;          // ceil((t0+16)/32) 32-wide s-blocks
  float mrun[8], lrun[8];
  #pragma unroll
  for (int v = 0; v < 8; ++v) { mrun[v] = -3e38f; lrun[v] = 0.f; }

  for (int blk = wave; blk < nblk; blk += 8) {
    const int s0 = blk * 32;
    v8f sc[2] = {z8, z8};
    for (int k0 = 0; k0 < C_; k0 += 32) {
      Frag a;
      const u32v4* qp = (const u32v4*)(sh_q + l15 * QROW + k0 + 8 * lsel);
      a.q[0] = qp[0];
      a.q[1] = qp[2];
      #pragma unroll
      for (int f = 0; f < 2; ++f) {
        Frag kf;
        const u32v4* kp = (const u32v4*)(kb + (size_t)(b * T_ + s0 + 16 * f + l15) * C_ + k0 + 8 * lsel);
        kf.q[0] = kp[0];
        kf.q[1] = kp[2];
        sc[f] = __builtin_amdgcn_wmma_f32_16x16x32_bf16(false, a.b, false, kf.b,
                                                        (short)0, sc[f], false, false);
      }
    }
    #pragma unroll
    for (int f = 0; f < 2; ++f) {
      const int sg = s0 + 16 * f + l15;       // D layout: N = l15, M = v + 8*lsel
      float sv[8];
      #pragma unroll
      for (int v = 0; v < 8; ++v) {
        float d = sc[f][v] * scale;
        const int tg = t0 + 8 * lsel + v;
        d = (sg <= tg) ? d : 0.0f;            // causal: outside tril -> 0
        sv[v] = (d == 0.0f) ? -1e30f : d;     // reference quirk: w==0 -> -inf
      }
      #pragma unroll
      for (int v = 0; v < 8; ++v) {           // online (m,l) update, rows via shfl
        float tm = hmax16(sv[v]);
        float mn = fmaxf(mrun[v], tm);
        float ts = hsum16(__expf(sv[v] - mn));
        lrun[v] = lrun[v] * __expf(mrun[v] - mn) + ts;
        mrun[v] = mn;
      }
      #pragma unroll
      for (int v = 0; v < 8; ++v)             // raw scores to padded LDS (bf16)
        sh_p[(size_t)(8 * lsel + v) * PROW + sg] = f2bf(sv[v]);
    }
  }
  if (l15 == 0) {
    #pragma unroll
    for (int v = 0; v < 8; ++v) {
      sh_m[wave * 16 + 8 * lsel + v] = mrun[v];
      sh_l[wave * 16 + 8 * lsel + v] = lrun[v];
    }
  }
  __syncthreads();

  // ---- combine partial stats (flash combine) ----
  if (threadIdx.x < 16) {
    const int r = threadIdx.x;
    float mf = -3e38f;
    #pragma unroll
    for (int w = 0; w < 8; ++w) mf = fmaxf(mf, sh_m[w * 16 + r]);
    float l = 0.f;
    #pragma unroll
    for (int w = 0; w < 8; ++w) l += __expf(sh_m[w * 16 + r] - mf) * sh_l[w * 16 + r];
    sh_mf[r] = mf;
    sh_lr[r] = 1.0f / l;
  }
  __syncthreads();

  // ---- Stage B: O = softmax(S) @ V, each wave owns a 64-wide h-chunk ----
  const int h0  = wave * 64;
  const float m_l = sh_mf[l15];               // A-frag row == l15 for every lane
  const float rl  = sh_lr[l15];
  v8f o[4];
  #pragma unroll
  for (int f = 0; f < 4; ++f) o[f] = z8;

  for (int blk = 0; blk < nblk; ++blk) {
    const int s0 = blk * 32;
    const u32v4* pp = (const u32v4*)(sh_p + l15 * PROW + s0 + 8 * lsel);
    u32v4 r0 = pp[0], r1 = pp[2];
    Frag pA;
    #pragma unroll
    for (int p = 0; p < 4; ++p) {
      pA.q[0][p] = expw(r0[p], m_l, rl);      // P = exp(s - m) / l, bf16
      pA.q[1][p] = expw(r1[p], m_l, rl);
    }
    #pragma unroll
    for (int f = 0; f < 4; ++f) {
      Frag vf;
      const u32v4* vp = (const u32v4*)(vt + (((size_t)(b * H_ + h0 + 16 * f + l15)) << 11)
                                          + s0 + 8 * lsel);
      vf.q[0] = vp[0];
      vf.q[1] = vp[2];
      o[f] = __builtin_amdgcn_wmma_f32_16x16x32_bf16(false, pA.b, false, vf.b,
                                                     (short)0, o[f], false, false);
    }
  }

  float* ob = out + ((size_t)(b * T_ + t0 + 8 * lsel)) * H_;
  #pragma unroll
  for (int f = 0; f < 4; ++f)
    #pragma unroll
    for (int v = 0; v < 8; ++v)
      ob[(size_t)v * H_ + h0 + 16 * f + l15] = o[f][v];
}

// ---------------------------------------------------------------------------
// Host-side launcher
// ---------------------------------------------------------------------------
extern "C" void kernel_launch(void* const* d_in, const int* in_sizes, int n_in,
                              void* d_out, int out_size, void* d_ws, size_t ws_size,
                              hipStream_t stream) {
  (void)in_sizes; (void)n_in; (void)out_size; (void)ws_size;
  const float* x  = (const float*)d_in[0];
  const float* Wq = (const float*)d_in[1];
  const float* Wk = (const float*)d_in[2];
  const float* Wv = (const float*)d_in[3];
  float* out = (float*)d_out;

  unsigned short* wsu = (unsigned short*)d_ws;
  const size_t nxt = (size_t)B_ * T_ * C_;    // 8,388,608
  const size_t nw  = (size_t)H_ * C_;         // 262,144
  unsigned short* xb  = wsu;
  unsigned short* wqb = xb  + nxt;
  unsigned short* wkb = wqb + nw;
  unsigned short* wvb = wkb + nw;
  unsigned short* qbf = wvb + nw;
  unsigned short* kbf = qbf + nxt;
  unsigned short* vtf = kbf + nxt;

  cvt_f32_bf16<<<(int)((nxt + 255) / 256), 256, 0, stream>>>(x,  xb,  (int)nxt);
  cvt_f32_bf16<<<(int)((nw  + 255) / 256), 256, 0, stream>>>(Wq, wqb, (int)nw);
  cvt_f32_bf16<<<(int)((nw  + 255) / 256), 256, 0, stream>>>(Wk, wkb, (int)nw);
  cvt_f32_bf16<<<(int)((nw  + 255) / 256), 256, 0, stream>>>(Wv, wvb, (int)nw);

  qkv_wmma<<<dim3(256, 8, 3), 128, 0, stream>>>(xb, wqb, wkb, wvb, qbf, kbf, vtf);

  const int smem = 16 * QROW * 2 + 16 * PROW * 2 + (128 + 128 + 16 + 16) * 4; // 83,584 B
  attn_wmma<<<dim3(T_ / 16, B_), 256, smem, stream>>>(qbf, kbf, vtf, out);
}
